// LiteMLAND_88218628260013
// MI455X (gfx1250) — matrix operand order repaired and checked by reading, not compiled, not run on hardware
//
#include <hip/hip_runtime.h>

#define DEV __device__ __forceinline__

typedef __attribute__((ext_vector_type(16))) _Float16 v16h;
typedef __attribute__((ext_vector_type(8)))  _Float16 v8h;
typedef __attribute__((ext_vector_type(8)))  float    v8f;

constexpr int Bz = 16;      // batch
constexpr int C  = 256;     // channels
constexpr int N  = 4096;    // H*W
constexpr int CP = 272;     // C + 16 (ones-pad column, padded to x16)
constexpr float EPS = 1e-5f;

DEV v8f wmma_f16(v16h a, v16h b, v8f c) {
  // v_wmma_f32_16x16x32_f16: 8 args (neg_a, A, neg_b, B, c_mod, C, reuse_a, reuse_b)
  return __builtin_amdgcn_wmma_f32_16x16x32_f16(false, a, false, b, (short)0, c,
                                                false, false);
}

DEV v8f vzero() { v8f z = {0.f,0.f,0.f,0.f,0.f,0.f,0.f,0.f}; return z; }

// A-operand (16x32 f16): lane holds row m; halves are A[m][k0+8h+e] and A[m][k0+16+8h+e].
// p must already point at  rowbase + k0 + 8*h  (16B aligned).
DEV v16h mk_a(const _Float16* p) {
  v8h x0 = *(const v8h*)p;
  v8h x1 = *(const v8h*)(p + 16);
  v16h a;
#pragma unroll
  for (int e = 0; e < 8; ++e) { a[e] = x0[e]; a[8 + e] = x1[e]; }
  return a;
}

// B-operand (32x16 f16): lane holds column n; halves are B[k0+8h+e][n] and B[k0+16+8h+e][n], read from
// the transposed (col-major-of-B == row "n", k contiguous) buffer.
// p must already point at  colbase + k0 + 8*h  (16B aligned).
DEV v16h mk_b(const _Float16* p) {
  v8h x0 = *(const v8h*)p;
  v8h x1 = *(const v8h*)(p + 16);
  v16h b;
#pragma unroll
  for (int e = 0; e < 8; ++e) { b[e] = x0[e]; b[8 + e] = x1[e]; }
  return b;
}

// -------------------------------------------------------------------------
// Kernel: zero f32 buffer
__global__ void k_zero(float* __restrict__ p, int n) {
  int i = blockIdx.x * 256 + threadIdx.x;
  if (i < n) p[i] = 0.0f;
}

// Kernel: f32 -> f16 convert
__global__ void k_cvt(const float* __restrict__ s, _Float16* __restrict__ d, int n) {
  int i = blockIdx.x * 256 + threadIdx.x;
  if (i < n) d[i] = (_Float16)s[i];
}

// -------------------------------------------------------------------------
// Kernel 1: qkv = w_qkv @ x  (per batch), interleaved split o = 3*c + j,
// ReLU on q,k.  Q -> (B,N,C) f16 ; K -> (B,C,N) f16 ; V -> (B,CP,N) f16
// (V row 256 = ones, rows 257..271 = zeros).
__global__ __launch_bounds__(256) void k_qkv(const float* __restrict__ x,
                                             const float* __restrict__ wqkv,
                                             _Float16* __restrict__ qh,
                                             _Float16* __restrict__ kh,
                                             _Float16* __restrict__ vh) {
  constexpr int XTS = C + 8;                 // padded LDS row stride (halves)
  __shared__ _Float16 xt[64 * XTS];          // x tile transposed: xt[nn][c]
  const int bb = blockIdx.y;
  const int n0 = blockIdx.x * 64;
  const int tid = threadIdx.x;

  { // stage x[b, c=tid, n0..n0+63] -> xt[nn][tid] as f16
    const float* src = x + ((size_t)bb * C + tid) * N + n0;
#pragma unroll 8
    for (int nn = 0; nn < 64; ++nn)
      xt[nn * XTS + tid] = (_Float16)src[nn];
  }
  { // fill the V pad rows (w = 256..271) for this n-block
#pragma unroll
    for (int i = 0; i < 4; ++i) {
      int idx = tid * 4 + i;
      int w = 256 + (idx >> 6), nn = idx & 63;
      vh[((size_t)bb * CP + w) * N + n0 + nn] =
          (w == 256) ? (_Float16)1.0f : (_Float16)0.0f;
    }
  }
  __syncthreads();

  const int wave = tid >> 5, lane = tid & 31;
  const int l15 = lane & 15, h = lane >> 4;

  for (int t = wave; t < 48; t += 8) {       // 48 output tiles: (j, c-out tile)
    const int j = t / 16;                    // 0=q, 1=k, 2=v
    const int c0 = (t % 16) * 16;
    const int arow = 3 * (c0 + l15) + j;     // interleaved w_qkv row
    v8f acc[4];
#pragma unroll
    for (int s = 0; s < 4; ++s) acc[s] = vzero();

    for (int k0 = 0; k0 < C; k0 += 32) {
      // A from global f32 weights, converted on the fly
      const float* ap = wqkv + (size_t)arow * C + k0 + 8 * h;
      v16h a;
#pragma unroll
      for (int e = 0; e < 8; ++e) {
        a[e]     = (_Float16)ap[e];
        a[8 + e] = (_Float16)ap[16 + e];
      }
#pragma unroll
      for (int s = 0; s < 4; ++s) {
        v16h bm = mk_b(&xt[(s * 16 + l15) * XTS + k0 + 8 * h]);
        acc[s] = wmma_f16(a, bm, acc[s]);
      }
    }
    // D element (r, lane): row o-rel = r + 8h, col n-rel = l15 (within subtile s)
    if (j == 0) {
#pragma unroll
      for (int s = 0; s < 4; ++s)
#pragma unroll
        for (int r = 0; r < 8; ++r)
          qh[((size_t)bb * N + n0 + s * 16 + l15) * C + c0 + r + 8 * h] =
              (_Float16)fmaxf(acc[s][r], 0.0f);
    } else if (j == 1) {
#pragma unroll
      for (int s = 0; s < 4; ++s)
#pragma unroll
        for (int r = 0; r < 8; ++r)
          kh[((size_t)bb * C + c0 + r + 8 * h) * N + n0 + s * 16 + l15] =
              (_Float16)fmaxf(acc[s][r], 0.0f);
    } else {
#pragma unroll
      for (int s = 0; s < 4; ++s)
#pragma unroll
        for (int r = 0; r < 8; ++r)
          vh[((size_t)bb * CP + c0 + r + 8 * h) * N + n0 + s * 16 + l15] =
              (_Float16)acc[s][r];
    }
  }
}

// -------------------------------------------------------------------------
// Kernel 2: kv[c,w] = sum_n K[n,c]*V[n,w]  as kvT[w][c], split-K over n with
// f32 atomics.  K stored (C,N), V stored (CP,N): both k(=n)-contiguous.
__global__ __launch_bounds__(256) void k_kv(const _Float16* __restrict__ kh,
                                            const _Float16* __restrict__ vh,
                                            float* __restrict__ kvf) {
  constexpr int NC = 128, NCS = NC + 8;
  extern __shared__ _Float16 sm2[];
  _Float16* kt = sm2;                  // [C ][NCS]
  _Float16* vt = sm2 + C * NCS;        // [CP][NCS]
  const int bb = blockIdx.y;
  const int n0 = blockIdx.x * NC;
  const int tid = threadIdx.x;

  { // stage K chunk
    const v8h* s = (const v8h*)(kh + ((size_t)bb * C + tid) * N + n0);
    v8h* d = (v8h*)(kt + tid * NCS);
#pragma unroll
    for (int i = 0; i < NC / 8; ++i) d[i] = s[i];
  }
  for (int rr = tid; rr < CP; rr += 256) { // stage V chunk
    const v8h* s = (const v8h*)(vh + ((size_t)bb * CP + rr) * N + n0);
    v8h* d = (v8h*)(vt + rr * NCS);
#pragma unroll
    for (int i = 0; i < NC / 8; ++i) d[i] = s[i];
  }
  __syncthreads();

  const int wave = tid >> 5, lane = tid & 31, l15 = lane & 31 & 15, h = lane >> 4;
  for (int t = wave; t < 16 * 17; t += 8) {
    const int ct = (t & 15) * 16;            // c tile base
    const int wt = (t >> 4) * 16;            // w tile base (0..256)
    v8f acc = vzero();
#pragma unroll
    for (int k0 = 0; k0 < NC; k0 += 32) {
      v16h a = mk_a(kt + (ct + l15) * NCS + k0 + 8 * h);
      v16h b = mk_b(vt + (wt + l15) * NCS + k0 + 8 * h);
      acc = wmma_f16(a, b, acc);
    }
#pragma unroll
    for (int r = 0; r < 8; ++r)
      unsafeAtomicAdd(&kvf[((size_t)bb * CP + wt + l15) * C + ct + r + 8 * h],
                      acc[r]);
  }
}

// -------------------------------------------------------------------------
// Kernel 3: out[n,w] = sum_c Q[n,c]*kv[c,w]; normalize by (out[n,256]+EPS);
// write out_norm (B,N,C) f16.
__global__ __launch_bounds__(256) void k_attn(const _Float16* __restrict__ qh,
                                              const _Float16* __restrict__ kvh,
                                              _Float16* __restrict__ onh) {
  constexpr int KS = C + 8;
  extern __shared__ _Float16 kvs[];    // [CP][KS]  (kvT: w-major, c contiguous)
  const int bb = blockIdx.y;
  const int tid = threadIdx.x;
  for (int rr = tid; rr < CP; rr += 256) {
    const v8h* s = (const v8h*)(kvh + ((size_t)bb * CP + rr) * C);
    v8h* d = (v8h*)(kvs + rr * KS);
#pragma unroll
    for (int i = 0; i < C / 8; ++i) d[i] = s[i];
  }
  __syncthreads();

  const int wave = tid >> 5, lane = tid & 31, l15 = lane & 15, h = lane >> 4;
  const int nt = blockIdx.x * 128 + wave * 16;
  const _Float16* qrow = qh + ((size_t)bb * N + nt + l15) * C;

  v16h areg[8];                         // Q rows cached across all 17 w-tiles
#pragma unroll
  for (int i = 0; i < 8; ++i) areg[i] = mk_a(qrow + i * 32 + 8 * h);

  // denominator tile (w = 256..271; col 0 of the tile is the real denominator)
  v8f accd = vzero();
#pragma unroll
  for (int i = 0; i < 8; ++i)
    accd = wmma_f16(areg[i], mk_b(kvs + (256 + l15) * KS + i * 32 + 8 * h), accd);

  float dv[8];
#pragma unroll
  for (int r = 0; r < 8; ++r)           // broadcast den for row m=r+8h from lane 16h
    dv[r] = __shfl(accd[r], h * 16, 32) + EPS;

  for (int wt = 0; wt < 16; ++wt) {
    v8f acc = vzero();
#pragma unroll
    for (int i = 0; i < 8; ++i)
      acc = wmma_f16(areg[i], mk_b(kvs + (wt * 16 + l15) * KS + i * 32 + 8 * h), acc);
#pragma unroll
    for (int r = 0; r < 8; ++r)
      onh[((size_t)bb * N + nt + r + 8 * h) * C + wt * 16 + l15] =
          (_Float16)(acc[r] / dv[r]);
  }
}

// -------------------------------------------------------------------------
// Kernel 4: y = w_proj @ out_norm^T, fused BatchNorm, f32 output (B,C,N).
__global__ __launch_bounds__(256) void k_proj(const _Float16* __restrict__ onh,
                                              const float* __restrict__ wproj,
                                              const float* __restrict__ gamma,
                                              const float* __restrict__ beta,
                                              const float* __restrict__ mean,
                                              const float* __restrict__ var,
                                              float* __restrict__ out) {
  constexpr int WS = C + 8;
  extern __shared__ char sm4[];
  _Float16* wp = (_Float16*)sm4;                 // [C][WS] f16 weights
  float* sc = (float*)(sm4 + (size_t)C * WS * 2);
  float* bc = sc + C;
  const int bb = blockIdx.y, n0 = blockIdx.x * 64, tid = threadIdx.x;

  { // stage w_proj row tid (f32->f16) + BN affine constants
    const float* s = wproj + (size_t)tid * C;
    _Float16* d = wp + tid * WS;
#pragma unroll 8
    for (int c = 0; c < C; ++c) d[c] = (_Float16)s[c];
    float v = gamma[tid] * rsqrtf(var[tid] + 1e-5f);
    sc[tid] = v;
    bc[tid] = beta[tid] - mean[tid] * v;
  }
  __syncthreads();

  const int wave = tid >> 5, lane = tid & 31, l15 = lane & 15, h = lane >> 4;
#pragma unroll
  for (int i = 0; i < 8; ++i) {
    const int t = wave * 8 + i;
    const int ot = (t >> 2) * 16;              // out-channel tile
    const int ns = (t & 3) * 16;               // n subtile
    const _Float16* arow = wp + (ot + l15) * WS;
    const _Float16* brow = onh + ((size_t)bb * N + n0 + ns + l15) * C;
    v8f acc = vzero();
#pragma unroll
    for (int k0 = 0; k0 < C; k0 += 32)
      acc = wmma_f16(mk_a(arow + k0 + 8 * h), mk_b(brow + k0 + 8 * h), acc);
#pragma unroll
    for (int r = 0; r < 8; ++r) {
      const int o = ot + r + 8 * h;
      out[((size_t)bb * C + o) * N + n0 + ns + l15] = acc[r] * sc[o] + bc[o];
    }
  }
}

// -------------------------------------------------------------------------
extern "C" void kernel_launch(void* const* d_in, const int* in_sizes, int n_in,
                              void* d_out, int out_size, void* d_ws, size_t ws_size,
                              hipStream_t stream) {
  const float* x     = (const float*)d_in[0];
  const float* wqkv  = (const float*)d_in[1];
  const float* wproj = (const float*)d_in[2];
  const float* gamma = (const float*)d_in[3];
  const float* beta  = (const float*)d_in[4];
  const float* mean  = (const float*)d_in[5];
  const float* var   = (const float*)d_in[6];
  float* out = (float*)d_out;

  _Float16* qh = (_Float16*)d_ws;                       // (B,N,C)
  _Float16* kh = qh + (size_t)Bz * N * C;               // (B,C,N)
  _Float16* vh = kh + (size_t)Bz * C * N;               // (B,CP,N)
  float*   kvf = (float*)(vh + (size_t)Bz * CP * N);    // (B,CP,C) f32 accum
  _Float16* kvh = (_Float16*)(kvf + (size_t)Bz * CP * C);
  _Float16* onh = kh;                                   // reuse K region (dead after k_kv)

  const int kvtot = Bz * CP * C;
  k_zero<<<(kvtot + 255) / 256, 256, 0, stream>>>(kvf, kvtot);
  k_qkv<<<dim3(N / 64, Bz), 256, 0, stream>>>(x, wqkv, qh, kh, vh);
  const size_t lds2 = (size_t)(C + CP) * 136 * 2;       // 143,616 B
  k_kv<<<dim3(N / 128, Bz), 256, lds2, stream>>>(kh, vh, kvf);
  k_cvt<<<(kvtot + 255) / 256, 256, 0, stream>>>(kvf, kvh, kvtot);
  const size_t lds3 = (size_t)CP * (C + 8) * 2;         // 143,616 B
  k_attn<<<dim3(N / 128, Bz), 256, lds3, stream>>>(qh, kvh, onh);
  const size_t lds4 = (size_t)C * (C + 8) * 2 + 2 * C * 4;
  k_proj<<<dim3(N / 64, Bz), 256, lds4, stream>>>(onh, wproj, gamma, beta, mean,
                                                  var, out);
}
